// PhysNet_56848187130524
// MI455X (gfx1250) — matrix operand order repaired and checked by reading, not compiled, not run on hardware
//
#include <hip/hip_runtime.h>
#include <math.h>

#define Bb    16
#define Aa    384
#define Nn    64
#define Ff    128
#define Kk    32
#define Ll    6
#define ALPHA 0.95f
#define ROWS  (Bb*Aa)          // 6144
#define AST   136              // padded LDS row stride (halfs) for GEMM tiles
#define FST   40               // padded LDS row stride (halfs) for conv tiles

typedef __attribute__((ext_vector_type(4)))  _Float16 hh4;
typedef __attribute__((ext_vector_type(8)))  _Float16 half8;
typedef __attribute__((ext_vector_type(16))) _Float16 v16h;
typedef __attribute__((ext_vector_type(8)))  float    v8f;

union H16 { v16h v; half8 h[2]; };

// Fast silu: hardware v_rcp_f32 instead of IEEE-exact division (output is
// rounded to f16 immediately, so rcp precision is more than sufficient).
__device__ __forceinline__ float silu_f(float x) {
    return x * __builtin_amdgcn_rcpf(1.0f + __expf(-x));
}

// ---------------------------------------------------------------------------
// Embedding gather: x[row, :] = embedding[Z[row], :]  (float4 vectorized)
// ---------------------------------------------------------------------------
__global__ void embed_kernel(const int* __restrict__ Z,
                             const float* __restrict__ emb,
                             float* __restrict__ X) {
    int row = blockIdx.x;
    int f4  = threadIdx.x * 4;
    const float4 v = *(const float4*)&emb[(long)Z[row] * Ff + f4];
    *(float4*)&X[(long)row * Ff + f4] = v;
}

// ---------------------------------------------------------------------------
// LDS staging helpers (inlined so address-space inference keeps ds_* ops)
// ---------------------------------------------------------------------------
// silu(X1 [+ X2]) -> f16 activation tile (rows r0..r0+127)
static __device__ __forceinline__
void stage_act(const float* __restrict__ X1, const float* __restrict__ X2,
               _Float16* __restrict__ A, long r0, int tid)
{
    #pragma unroll 4
    for (int e = 0; e < 16; ++e) {
        int idx = tid + 256 * e;          // 0..4095 float4s
        int m   = idx >> 5;               // 0..127
        int k4  = (idx & 31) * 4;
        long off = (r0 + m) * Ff + k4;
        float4 v = *(const float4*)&X1[off];
        if (X2) {
            float4 u = *(const float4*)&X2[off];
            v.x += u.x; v.y += u.y; v.z += u.z; v.w += u.w;
        }
        hh4 hv = { (_Float16)silu_f(v.x), (_Float16)silu_f(v.y),
                   (_Float16)silu_f(v.z), (_Float16)silu_f(v.w) };
        *(hh4*)&A[m * AST + k4] = hv;
    }
}

// Weight matrix [k][n] row-major f32 -> transposed f16 LDS tile Wt[n][k]
static __device__ __forceinline__
void stage_W(const float* __restrict__ W, _Float16* __restrict__ Wt, int tid)
{
    #pragma unroll 4
    for (int e = 0; e < 16; ++e) {
        int idx = tid + 256 * e;
        int k   = idx >> 5;
        int n4  = (idx & 31) * 4;
        float4 w = *(const float4*)&W[(long)k * Ff + n4];
        Wt[(n4 + 0) * AST + k] = (_Float16)w.x;
        Wt[(n4 + 1) * AST + k] = (_Float16)w.y;
        Wt[(n4 + 2) * AST + k] = (_Float16)w.z;
        Wt[(n4 + 3) * AST + k] = (_Float16)w.w;
    }
}

static __device__ __forceinline__
void stage_b(const float* __restrict__ bias, float* __restrict__ bsh, int tid)
{
    if (tid < 128) bsh[tid] = bias[tid];
}

// ---------------------------------------------------------------------------
// One fused matmul step:  y = wmma(Asrc, Wt) + bias [+ Add1][+ Add2]
//   - optional raw f32 store of y to Gout
//   - optional f16 silu(y) store into the next LDS activation tile Adst
// Inlined at every call site so the nullable paths constant-fold away.
// ---------------------------------------------------------------------------
static __device__ __forceinline__
void mm_step(const _Float16* __restrict__ Asrc, const _Float16* __restrict__ Wt,
             const float* __restrict__ bsh,
             const float* __restrict__ Add1, const float* __restrict__ Add2,
             float* __restrict__ Gout, _Float16* __restrict__ Adst,
             long r0, int tid)
{
    const int wave = tid >> 5;
    const int lane = tid & 31;
    const int hi   = (lane >= 16) ? 1 : 0;
    const int lrow = lane & 15;
    const int m0   = wave * 16;

    // A-fragments for this wave's 16-row band (ISA 16-bit A layout: per lane,
    // K chunks [k0+hi*8,+8) and [k0+16+hi*8,+8)).
    H16 afrag[4];
    #pragma unroll
    for (int kb = 0; kb < 4; ++kb) {
        int k0 = kb * 32;
        afrag[kb].h[0] = *(const half8*)&Asrc[(m0 + lrow) * AST + k0 + hi * 8];
        afrag[kb].h[1] = *(const half8*)&Asrc[(m0 + lrow) * AST + k0 + 16 + hi * 8];
    }

    for (int nb = 0; nb < 8; ++nb) {
        const int n0 = nb * 16;
        v8f acc = {};
        #pragma unroll
        for (int kb = 0; kb < 4; ++kb) {
            int k0 = kb * 32;
            // B layout: lane = column, K chunks [k0+hi*8,+8) and [k0+16+hi*8,+8).
            H16 bfrag;
            const _Float16* q = &Wt[(n0 + lrow) * AST + k0 + hi * 8];
            bfrag.h[0] = *(const half8*)&q[0];
            bfrag.h[1] = *(const half8*)&q[16];
            acc = __builtin_amdgcn_wmma_f32_16x16x32_f16(
                      false, afrag[kb].v, false, bfrag.v,
                      (short)0, acc, false, false);
        }
        // Epilogue: C layout = 8 rows per lane, col = n0 + lane%16.
        const int col   = n0 + lrow;
        const float bb  = bsh[col];
        const long base = (r0 + m0 + hi * 8) * Ff + col;

        float a1v[8], a2v[8];
        if (Add1) {
            #pragma unroll
            for (int r = 0; r < 8; ++r) a1v[r] = Add1[base + (long)r * Ff];
        }
        if (Add2) {
            #pragma unroll
            for (int r = 0; r < 8; ++r) a2v[r] = Add2[base + (long)r * Ff];
        }
        #pragma unroll
        for (int r = 0; r < 8; ++r) {
            float y = acc[r] + bb;
            if (Add1) y += a1v[r];
            if (Add2) y += a2v[r];
            if (Gout) Gout[base + (long)r * Ff] = y;
            if (Adst) Adst[(m0 + r + hi * 8) * AST + col] = (_Float16)silu_f(y);
        }
    }
}

// ---------------------------------------------------------------------------
// Stage A (row-local half of a layer): 8 chained matmuls entirely in LDS.
//   x1 = residual(x)                      (steps 0-1)
//   xi = dense0(residual(x1)) -> bi       (steps 2-4)
//   xj = dense1(residual(x1)) -> bj       (steps 5-7)
// ---------------------------------------------------------------------------
struct SAArgs {
    const float* X;            // current x
    const float* W[8];         // W1_0,W2_0, W1_1,W2_1, DW0, W1_2,W2_2, DW1
    const float* Bp[8];        // matching biases
    float* x1;
    float* bi;
    float* bj;
};

__global__ __launch_bounds__(256)
void stageA_kernel(SAArgs a)
{
    __shared__ __align__(16) _Float16 A0[128 * AST];
    __shared__ __align__(16) _Float16 A1[128 * AST];
    __shared__ __align__(16) _Float16 A2[128 * AST];
    __shared__ __align__(16) _Float16 Wt[128 * AST];
    __shared__ float bsh[128];

    const int  tid = threadIdx.x;
    const long r0  = (long)blockIdx.x * 128;

    stage_act(a.X, nullptr, A0, r0, tid);
    stage_W(a.W[0], Wt, tid); stage_b(a.Bp[0], bsh, tid);
    __syncthreads();
    mm_step(A0, Wt, bsh, nullptr, nullptr, nullptr, A1, r0, tid);       // t0
    __syncthreads();
    stage_W(a.W[1], Wt, tid); stage_b(a.Bp[1], bsh, tid);
    __syncthreads();
    mm_step(A1, Wt, bsh, a.X, nullptr, a.x1, A2, r0, tid);              // x1 (+silu->A2)
    __syncthreads();
    stage_W(a.W[2], Wt, tid); stage_b(a.Bp[2], bsh, tid);
    __syncthreads();
    mm_step(A2, Wt, bsh, nullptr, nullptr, nullptr, A1, r0, tid);       // t1
    __syncthreads();
    stage_W(a.W[3], Wt, tid); stage_b(a.Bp[3], bsh, tid);
    __syncthreads();
    mm_step(A1, Wt, bsh, a.x1, nullptr, nullptr, A0, r0, tid);          // xi_res
    __syncthreads();
    stage_W(a.W[4], Wt, tid); stage_b(a.Bp[4], bsh, tid);
    __syncthreads();
    mm_step(A0, Wt, bsh, nullptr, nullptr, a.bi, nullptr, r0, tid);     // bi = dense0
    __syncthreads();
    stage_W(a.W[5], Wt, tid); stage_b(a.Bp[5], bsh, tid);
    __syncthreads();
    mm_step(A2, Wt, bsh, nullptr, nullptr, nullptr, A1, r0, tid);       // t2 (from silu(x1))
    __syncthreads();
    stage_W(a.W[6], Wt, tid); stage_b(a.Bp[6], bsh, tid);
    __syncthreads();
    mm_step(A1, Wt, bsh, a.x1, nullptr, nullptr, A0, r0, tid);          // xj_res
    __syncthreads();
    stage_W(a.W[7], Wt, tid); stage_b(a.Bp[7], bsh, tid);
    __syncthreads();
    mm_step(A0, Wt, bsh, nullptr, nullptr, a.bj, nullptr, r0, tid);     // bj = dense1
}

// ---------------------------------------------------------------------------
// Stage C (after cfconv): 5 chained matmuls.
//   v   = dense2(residual(xi + cv))       (steps 0-2, s = x1+v spilled)
//   x'  = x + residual(x1 + v)            (steps 3-4)
// ---------------------------------------------------------------------------
struct SCArgs {
    const float* bi;
    const float* cv;
    const float* x1;
    const float* X;            // current x
    const float* W[5];         // W1_3,W2_3, DW2, W1_4,W2_4
    const float* Bp[5];
    float* s;
    float* out;
};

__global__ __launch_bounds__(256)
void stageC_kernel(SCArgs a)
{
    __shared__ __align__(16) _Float16 A0[128 * AST];
    __shared__ __align__(16) _Float16 A1[128 * AST];
    __shared__ __align__(16) _Float16 Wt[128 * AST];
    __shared__ float bsh[128];

    const int  tid = threadIdx.x;
    const long r0  = (long)blockIdx.x * 128;

    stage_act(a.bi, a.cv, A0, r0, tid);                                  // silu(xi+cv)
    stage_W(a.W[0], Wt, tid); stage_b(a.Bp[0], bsh, tid);
    __syncthreads();
    mm_step(A0, Wt, bsh, nullptr, nullptr, nullptr, A1, r0, tid);        // t3
    __syncthreads();
    stage_W(a.W[1], Wt, tid); stage_b(a.Bp[1], bsh, tid);
    __syncthreads();
    mm_step(A1, Wt, bsh, a.bi, a.cv, nullptr, A0, r0, tid);              // v_res
    __syncthreads();
    stage_W(a.W[2], Wt, tid); stage_b(a.Bp[2], bsh, tid);
    __syncthreads();
    mm_step(A0, Wt, bsh, a.x1, nullptr, a.s, A1, r0, tid);               // s = x1 + dense2
    __syncthreads();
    stage_W(a.W[3], Wt, tid); stage_b(a.Bp[3], bsh, tid);
    __syncthreads();
    mm_step(A1, Wt, bsh, nullptr, nullptr, nullptr, A0, r0, tid);        // t4
    __syncthreads();
    stage_W(a.W[4], Wt, tid); stage_b(a.Bp[4], bsh, tid);
    __syncthreads();
    mm_step(A0, Wt, bsh, a.s, a.X, a.out, nullptr, r0, tid);             // x_new
}

// ---------------------------------------------------------------------------
// cfconv: one workgroup per (b,a) atom.
//   - recompute Bernstein RBF f_ij[64][32] in LDS (f16, mask folded in)
//   - Wf = f_ij @ filtW_l via WMMA (wave w owns feature cols 16w..16w+15)
//   - multiply by gathered neighbor features, reduce over N (regs + shfl_xor)
// ---------------------------------------------------------------------------
__global__ __launch_bounds__(256)
void conv_kernel(const float* __restrict__ positions,
                 const int*   __restrict__ neighbors,
                 const float* __restrict__ mask,
                 const float* __restrict__ filtW_l,   // [K, F]
                 const float* __restrict__ XJ,        // [ROWS, F]
                 float* __restrict__ OutCV)           // [ROWS, F]
{
    __shared__ __align__(16) _Float16 fij[Nn * FST];
    __shared__ __align__(16) _Float16 Ft [Ff * FST];
    __shared__ int   nbsh[Nn];
    __shared__ float xssh[Nn], lgsh[Nn], msh[Nn];
    __shared__ float logcsh[Kk];

    const int tid = threadIdx.x;
    const int ba  = blockIdx.x;
    const int b   = ba / Aa;

    if (tid < Nn) {
        int   j  = neighbors[(long)ba * Nn + tid];
        float mk = mask[(long)ba * Nn + tid];
        const float* pi = &positions[(long)ba * 3];
        const float* pj = &positions[((long)b * Aa + j) * 3];
        float dx = pj[0] - pi[0], dy = pj[1] - pi[1], dz = pj[2] - pi[2];
        float d2 = dx * dx + dy * dy + dz * dz;
        float r  = ((mk > 0.f) ? sqrtf(d2) : 1.0f) * mk;
        float x  = -ALPHA * r;
        float xs = (mk > 0.f) ? x : -1.0f;
        float xm = fminf(xs, -1e-6f);
        nbsh[tid] = j;
        xssh[tid] = xs;
        lgsh[tid] = logf(-expm1f(xm));   // log(1 - e^x)
        msh[tid]  = mk;
    }
    if (tid < Kk) {
        float kf = (float)tid;
        logcsh[tid] = lgammaf((float)Kk) - lgammaf(kf + 1.0f) - lgammaf((float)Kk - kf);
    }
    __syncthreads();

    // Bernstein expansion with exp-weighting, mask folded, f16.
    for (int idx = tid; idx < Nn * Kk; idx += 256) {
        int n = idx >> 5, k = idx & 31;
        float xs = xssh[n];
        float v  = __expf(logcsh[k] + (float)k * xs + (float)(Kk - 1 - k) * lgsh[n] + xs) * msh[n];
        fij[n * FST + k] = (_Float16)v;
    }
    // Filter transposed into LDS: Ft[f][k]  (float4 coalesced over f).
    for (int idx = tid; idx < (Kk * Ff) / 4; idx += 256) {
        int k  = idx >> 5;             // 0..31
        int f4 = (idx & 31) * 4;       // 0..124
        float4 w = *(const float4*)&filtW_l[(long)k * Ff + f4];
        Ft[(f4 + 0) * FST + k] = (_Float16)w.x;
        Ft[(f4 + 1) * FST + k] = (_Float16)w.y;
        Ft[(f4 + 2) * FST + k] = (_Float16)w.z;
        Ft[(f4 + 3) * FST + k] = (_Float16)w.w;
    }
    __syncthreads();

    const int wave = tid >> 5;
    const int lane = tid & 31;
    const int hi   = (lane >= 16) ? 1 : 0;
    const int lrow = lane & 15;
    const int col  = wave * 16 + lrow;     // feature column this lane owns

    // B fragment (K=32 x 16 cols) from Ft.
    H16 bfrag;
    const _Float16* q = &Ft[col * FST + hi * 8];
    bfrag.h[0] = *(const half8*)&q[0];
    bfrag.h[1] = *(const half8*)&q[16];

    float partial = 0.f;
    #pragma unroll
    for (int m = 0; m < 4; ++m) {
        H16 afrag;
        int nrow = m * 16 + lrow;
        afrag.h[0] = *(const half8*)&fij[nrow * FST + hi * 8];
        afrag.h[1] = *(const half8*)&fij[nrow * FST + 16 + hi * 8];
        v8f acc = {};
        acc = __builtin_amdgcn_wmma_f32_16x16x32_f16(
                  false, afrag.v, false, bfrag.v, (short)0, acc, false, false);
        // Wf tile * gathered neighbor features, accumulate over rows (N dim).
        float yj[8];
        #pragma unroll
        for (int r = 0; r < 8; ++r) {
            int nn = m * 16 + r + hi * 8;
            yj[r] = XJ[((long)b * Aa + nbsh[nn]) * Ff + col];
        }
        #pragma unroll
        for (int r = 0; r < 8; ++r) partial += acc[r] * yj[r];
    }
    // lanes l and l^16 hold the two row-halves of the same feature column
    partial += __shfl_xor(partial, 16, 32);
    if (lane < 16) OutCV[(long)ba * Ff + col] = partial;
}

// ---------------------------------------------------------------------------
// Host orchestration: 1 embed + per layer {stageA, conv, stageC} = 19 launches
// ---------------------------------------------------------------------------
extern "C" void kernel_launch(void* const* d_in, const int* in_sizes, int n_in,
                              void* d_out, int out_size, void* d_ws, size_t ws_size,
                              hipStream_t stream)
{
    const int*   Z    = (const int*)d_in[0];
    const float* pos  = (const float*)d_in[1];
    const int*   nbr  = (const int*)d_in[2];
    const float* msk  = (const float*)d_in[3];
    const float* emb  = (const float*)d_in[4];
    const float* rW1  = (const float*)d_in[5];
    const float* rb1  = (const float*)d_in[6];
    const float* rW2  = (const float*)d_in[7];
    const float* rb2  = (const float*)d_in[8];
    const float* dW   = (const float*)d_in[9];
    const float* db   = (const float*)d_in[10];
    const float* fW   = (const float*)d_in[11];

    float* ws = (float*)d_ws;
    const size_t T = (size_t)ROWS * Ff;     // 786432 floats per tensor
    float* bufA = ws + 0 * T;
    float* bufB = ws + 1 * T;
    float* x1g  = ws + 2 * T;
    float* big  = ws + 3 * T;
    float* bjg  = ws + 4 * T;
    float* cvg  = ws + 5 * T;
    float* sg   = ws + 6 * T;

    float* cur = bufA;
    float* nxt = bufB;

    embed_kernel<<<ROWS, Ff / 4, 0, stream>>>(Z, emb, cur);

    dim3 gg(ROWS / 128), bb(256);
    for (int l = 0; l < Ll; ++l) {
        const float *W1[5], *B1[5], *W2[5], *B2[5];
        for (int i = 0; i < 5; ++i) {
            W1[i] = rW1 + ((size_t)(l * 5 + i)) * Ff * Ff;
            B1[i] = rb1 + ((size_t)(l * 5 + i)) * Ff;
            W2[i] = rW2 + ((size_t)(l * 5 + i)) * Ff * Ff;
            B2[i] = rb2 + ((size_t)(l * 5 + i)) * Ff;
        }
        const float *DWp[3], *DBp[3];
        for (int i = 0; i < 3; ++i) {
            DWp[i] = dW + ((size_t)(l * 3 + i)) * Ff * Ff;
            DBp[i] = db + ((size_t)(l * 3 + i)) * Ff;
        }
        const float* FL = fW + (size_t)l * Kk * Ff;

        SAArgs sa;
        sa.X = cur;
        sa.W[0] = W1[0]; sa.Bp[0] = B1[0];
        sa.W[1] = W2[0]; sa.Bp[1] = B2[0];
        sa.W[2] = W1[1]; sa.Bp[2] = B1[1];
        sa.W[3] = W2[1]; sa.Bp[3] = B2[1];
        sa.W[4] = DWp[0]; sa.Bp[4] = DBp[0];
        sa.W[5] = W1[2]; sa.Bp[5] = B1[2];
        sa.W[6] = W2[2]; sa.Bp[6] = B2[2];
        sa.W[7] = DWp[1]; sa.Bp[7] = DBp[1];
        sa.x1 = x1g; sa.bi = big; sa.bj = bjg;
        stageA_kernel<<<gg, bb, 0, stream>>>(sa);

        conv_kernel<<<ROWS, 256, 0, stream>>>(pos, nbr, msk, FL, bjg, cvg);

        SCArgs sc;
        sc.bi = big; sc.cv = cvg; sc.x1 = x1g; sc.X = cur;
        sc.W[0] = W1[3]; sc.Bp[0] = B1[3];
        sc.W[1] = W2[3]; sc.Bp[1] = B2[3];
        sc.W[2] = DWp[2]; sc.Bp[2] = DBp[2];
        sc.W[3] = W1[4]; sc.Bp[3] = B1[4];
        sc.W[4] = W2[4]; sc.Bp[4] = B2[4];
        sc.s = sg;
        sc.out = (l == Ll - 1) ? (float*)d_out : nxt;
        stageC_kernel<<<gg, bb, 0, stream>>>(sc);

        float* sw = cur; cur = nxt; nxt = sw;
    }
    (void)in_sizes; (void)n_in; (void)out_size; (void)ws_size;
}
